// Attr_Relation_Net_75797582840044
// MI455X (gfx1250) — matrix operand order repaired and checked by reading, not compiled, not run on hardware
//
#include <hip/hip_runtime.h>
#include <hip/hip_bf16.h>
#include <math.h>

typedef unsigned short ushort_t;
typedef unsigned int   uint_t;

typedef __attribute__((ext_vector_type(16))) __bf16 v16bf;
typedef __attribute__((ext_vector_type(8)))  float  v8f;

union Frag { v16bf v; uint4 q[2]; };

#define WAVES   4            // waves per block (128 threads)
#define LDX     72           // bufX row stride (elems): 144B, 16B aligned, conflict-free
#define LDY     264          // bufY row stride (elems): 528B, 16B aligned, conflict-free

static __device__ inline ushort_t f2bf(float f) {
    uint_t u = __float_as_uint(f);
    uint_t r = u + 0x7FFFu + ((u >> 16) & 1u);   // round-to-nearest-even
    return (ushort_t)(r >> 16);
}

// Branch-free GELU (tanh form rewritten as sigmoid):
// v_exp_f32 + single v_rcp_f32 (hardware rcp, not IEEE div sequence) + ~5 VALU.
static __device__ inline float gelu_f(float x) {
    float u = x * (1.5957691216f + 0.0713548162726f * x * x); // 2*0.79788456*(x+0.044715x^3)
    return x * __builtin_amdgcn_rcpf(1.0f + __expf(-u));
}

// ---- load all KT A-fragments ([16 x 32*KT] bf16 in LDS) into registers
template<int KT, int LDA>
static __device__ inline void load_afrags(const ushort_t* __restrict__ A, int lane,
                                          Frag* __restrict__ af) {
    const int m  = lane & 15;
    const int kb = (lane >> 4) * 8;
#pragma unroll
    for (int kt = 0; kt < KT; ++kt) {
        const ushort_t* ar = A + m * LDA + kt * 32;
        af[kt].q[0] = *(const uint4*)(ar + kb);        // K = kb .. kb+7
        af[kt].q[1] = *(const uint4*)(ar + kb + 16);   // K = kb+16 .. kb+23
    }
}

// ---- one 16x16 C-tile: register A-frags  x  global packed B-frags
template<int KT>
static __device__ inline v8f mm_frags(const Frag* __restrict__ af,
                                      const ushort_t* __restrict__ Btiles,
                                      int nt, int lane) {
    v8f acc = {};
#pragma unroll
    for (int kt = 0; kt < KT; ++kt) {
        Frag b;
        const ushort_t* br = Btiles + (size_t)(nt * KT + kt) * 512 + lane * 16;
        b.q[0] = *(const uint4*)(br);
        b.q[1] = *(const uint4*)(br + 8);
        acc = __builtin_amdgcn_wmma_f32_16x16x32_bf16(
                  false, af[kt].v, false, b.v, (short)0, acc, false, false);
    }
    return acc;
}

// ---------------- prep: Gram matrix G = feature_emb @ feature_emb^T (64x64 f32)
__global__ void gram_kernel(const float* __restrict__ fe, float* __restrict__ G) {
    int t = blockIdx.x * blockDim.x + threadIdx.x;
    if (t >= 64 * 64) return;
    int i = t >> 6, j = t & 63;
    const float* a = fe + i * 256;
    const float* b = fe + j * 256;
    float s = 0.f;
#pragma unroll 8
    for (int k = 0; k < 256; ++k) s += a[k] * b[k];
    G[t] = s;
}

// ---------------- prep: pack W[K][N] (f32 row-major) -> bf16 WMMA B-fragments
__global__ void pack_kernel(const float* __restrict__ W, ushort_t* __restrict__ out,
                            int K, int N) {
    int KT = K >> 5, NT = N >> 4;
    int t = blockIdx.x * blockDim.x + threadIdx.x;
    if (t >= KT * NT * 32) return;
    int lane = t & 31, tileI = t >> 5;
    int kt = tileI % KT, nt = tileI / KT;
    int n  = nt * 16 + (lane & 15);
    int k0 = kt * 32 + (lane >> 4) * 8;
    ushort_t* dst = out + (size_t)tileI * 512 + lane * 16;
#pragma unroll
    for (int i = 0; i < 16; ++i)
        dst[i] = f2bf(W[(size_t)(k0 + (i & 7) + ((i >> 3) << 4)) * N + n]);
}

// ---------------- main: one wave per 16-edge tile, fused chain
__global__ __launch_bounds__(32 * WAVES)
void attr_relation_main(const float* __restrict__ known_mask,
                        const int*   __restrict__ obs_idx,
                        const int*   __restrict__ obs_mask_idx,
                        const int*   __restrict__ attr_idx,
                        const float* __restrict__ obs_embs,
                        const float* __restrict__ G,
                        const ushort_t* __restrict__ Brm1, const float* __restrict__ rm_b1,
                        const ushort_t* __restrict__ Brm2, const float* __restrict__ rm_b2,
                        const ushort_t* __restrict__ Brr,  const float* __restrict__ rr_b,
                        const ushort_t* __restrict__ Brc,  const float* __restrict__ rc_b,
                        float* __restrict__ out) {
    __shared__ ushort_t bufX[WAVES][16][LDX];   // K=64 stages
    __shared__ ushort_t bufY[WAVES][16][LDY];   // K=256 stages
    __shared__ int attr_s[WAVES][16];
    __shared__ int oidx_s[WAVES][16];

    const int lane = threadIdx.x & 31;
    const int wid  = threadIdx.x >> 5;
    const int e0   = (blockIdx.x * WAVES + wid) * 16;   // E = 400000 = 25000*16, exact
    const int hi8  = (lane >> 4) * 8;
    const int ncol = lane & 15;

    // ---- stage 0: s = softmax(m_i * m_j) over F=64 -> bufX (bf16)
    {
        const int r = lane >> 1, half = lane & 1;
        const int e = e0 + r;
        const int mrow = obs_mask_idx[e];
        const int attr = attr_idx[e];
        if (half == 0) { attr_s[wid][r] = attr; oidx_s[wid][r] = obs_idx[e]; }
        float v[32];
        const float4* src = (const float4*)(known_mask + (size_t)mrow * 64 + half * 32);
#pragma unroll
        for (int i = 0; i < 8; ++i) {
            float4 q = src[i];
            v[4*i] = q.x; v[4*i+1] = q.y; v[4*i+2] = q.z; v[4*i+3] = q.w;
        }
        const int fbase = half * 32;
#pragma unroll
        for (int i = 0; i < 32; ++i) if (fbase + i == attr) v[i] = 0.f;   // m_j zero
        float mx = v[0];
#pragma unroll
        for (int i = 1; i < 32; ++i) mx = fmaxf(mx, v[i]);
        mx = fmaxf(mx, __shfl_xor(mx, 1, 32));
        float sum = 0.f;
#pragma unroll
        for (int i = 0; i < 32; ++i) { v[i] = __expf(v[i] - mx); sum += v[i]; }
        sum += __shfl_xor(sum, 1, 32);
        const float inv = __builtin_amdgcn_rcpf(sum);
        uint4* dst = (uint4*)&bufX[wid][r][fbase];
#pragma unroll
        for (int i = 0; i < 4; ++i) {
            uint4 p;
            p.x = (uint_t)f2bf(v[8*i+0]*inv) | ((uint_t)f2bf(v[8*i+1]*inv) << 16);
            p.y = (uint_t)f2bf(v[8*i+2]*inv) | ((uint_t)f2bf(v[8*i+3]*inv) << 16);
            p.z = (uint_t)f2bf(v[8*i+4]*inv) | ((uint_t)f2bf(v[8*i+5]*inv) << 16);
            p.w = (uint_t)f2bf(v[8*i+6]*inv) | ((uint_t)f2bf(v[8*i+7]*inv) << 16);
            dst[i] = p;
        }
    }

    Frag af2[2];   // A-fragments for K=64 stages
    Frag af8[8];   // A-fragments for K=256 stages

    // ---- stage 1: h1 = gelu(s @ rm_W1 + b1)        [16 x 256] -> bufY
    load_afrags<2, LDX>(&bufX[wid][0][0], lane, af2);
    for (int nt = 0; nt < 16; ++nt) {
        v8f acc = mm_frags<2>(af2, Brm1, nt, lane);
        const int n = nt * 16 + ncol;
        const float b = rm_b1[n];
#pragma unroll
        for (int j = 0; j < 8; ++j)
            bufY[wid][j + hi8][n] = f2bf(gelu_f(acc[j] + b));
    }

    // ---- stage 2: a = G[attr,:] * gelu(h1 @ rm_W2 + b2)   [16 x 64] -> bufX
    load_afrags<8, LDY>(&bufY[wid][0][0], lane, af8);
    for (int nt = 0; nt < 4; ++nt) {
        v8f acc = mm_frags<8>(af8, Brm2, nt, lane);
        const int n = nt * 16 + ncol;
        const float b = rm_b2[n];
#pragma unroll
        for (int j = 0; j < 8; ++j) {
            const int mr = j + hi8;
            const float g = gelu_f(acc[j] + b);
            bufX[wid][mr][n] = f2bf(G[attr_s[wid][mr] * 64 + n] * g);
        }
    }

    // ---- stage 3: z = obs_h * gelu(a @ rr_W + rr_b)       [16 x 256] -> bufY
    load_afrags<2, LDX>(&bufX[wid][0][0], lane, af2);
    for (int nt = 0; nt < 16; ++nt) {
        v8f acc = mm_frags<2>(af2, Brr, nt, lane);
        const int n = nt * 16 + ncol;
        const float b = rr_b[n];
#pragma unroll
        for (int j = 0; j < 8; ++j) {
            const int mr = j + hi8;
            const float g = gelu_f(acc[j] + b);
            bufY[wid][mr][n] = f2bf(obs_embs[(size_t)oidx_s[wid][mr] * 256 + n] * g);
        }
    }

    // ---- stage 4: out = gelu(z @ rc_W + rc_b)             [16 x 256] -> global (NT)
    load_afrags<8, LDY>(&bufY[wid][0][0], lane, af8);
    for (int nt = 0; nt < 16; ++nt) {
        v8f acc = mm_frags<8>(af8, Brc, nt, lane);
        const int n = nt * 16 + ncol;
        const float b = rc_b[n];
#pragma unroll
        for (int j = 0; j < 8; ++j)
            __builtin_nontemporal_store(gelu_f(acc[j] + b),
                                        out + (size_t)(e0 + j + hi8) * 256 + n);
    }
}

extern "C" void kernel_launch(void* const* d_in, const int* in_sizes, int n_in,
                              void* d_out, int out_size, void* d_ws, size_t ws_size,
                              hipStream_t stream) {
    const float* known_mask   = (const float*)d_in[0];
    const int*   obs_idx      = (const int*)  d_in[1];
    const int*   obs_mask_idx = (const int*)  d_in[2];
    const int*   attr_idx     = (const int*)  d_in[3];
    const float* obs_embs     = (const float*)d_in[4];
    const float* feature_emb  = (const float*)d_in[5];
    const float* rm_W1 = (const float*)d_in[6];
    const float* rm_b1 = (const float*)d_in[7];
    const float* rm_W2 = (const float*)d_in[8];
    const float* rm_b2 = (const float*)d_in[9];
    const float* rr_W  = (const float*)d_in[10];
    const float* rr_b  = (const float*)d_in[11];
    const float* rc_W  = (const float*)d_in[12];
    const float* rc_b  = (const float*)d_in[13];
    float* out = (float*)d_out;

    char* ws = (char*)d_ws;
    float*    G    = (float*)ws;                              // 64*64*4   = 16 KB
    ushort_t* Brm1 = (ushort_t*)(ws + 16384);                 // 32 tiles  = 32 KB
    ushort_t* Brm2 = (ushort_t*)(ws + 16384 + 32768);         // 32 tiles  = 32 KB
    ushort_t* Brr  = (ushort_t*)(ws + 16384 + 65536);         // 32 tiles  = 32 KB
    ushort_t* Brc  = (ushort_t*)(ws + 16384 + 98304);         // 128 tiles = 128 KB

    gram_kernel<<<16, 256, 0, stream>>>(feature_emb, G);
    pack_kernel<<<4, 256, 0, stream>>>(rm_W1, Brm1, 64, 256);   // KT=2 NT=16
    pack_kernel<<<4, 256, 0, stream>>>(rm_W2, Brm2, 256, 64);   // KT=8 NT=4
    pack_kernel<<<4, 256, 0, stream>>>(rr_W,  Brr,  64, 256);   // KT=2 NT=16
    pack_kernel<<<16, 256, 0, stream>>>(rc_W,  Brc,  256, 256); // KT=8 NT=16

    const int E = 400000;                 // 25000 wave-tiles of 16 edges
    const int tiles = E / 16;
    attr_relation_main<<<tiles / WAVES, 32 * WAVES, 0, stream>>>(
        known_mask, obs_idx, obs_mask_idx, attr_idx, obs_embs,
        G, Brm1, rm_b1, Brm2, rm_b2, Brr, rr_b, Brc, rc_b, out);
}
